// SelfAttentionLayer_84799834293007
// MI455X (gfx1250) — matrix operand order repaired and checked by reading, not compiled, not run on hardware
//
#include <hip/hip_runtime.h>
#include <hip/hip_bf16.h>

// Problem constants (match reference)
#define BB 32
#define NN 1024
#define CC 256
#define WW 256
#define NEGV (-1000000.0f)

typedef __attribute__((ext_vector_type(16))) __bf16 v16bf;
typedef __attribute__((ext_vector_type(8)))  __bf16 v8bf;
typedef __attribute__((ext_vector_type(8)))  float  v8f;

static __device__ __forceinline__ v8f wmma_bf16(v16bf a, v16bf b, v8f c) {
  // D = A(16x32 bf16) * B(32x16 bf16) + C(16x16 f32)
  return __builtin_amdgcn_wmma_f32_16x16x32_bf16(false, a, false, b, (short)0, c,
                                                 false, false);
}

static __device__ __forceinline__ v16bf cat8(v8bf lo, v8bf hi) {
  return __builtin_shufflevector(lo, hi, 0, 1, 2, 3, 4, 5, 6, 7, 8, 9, 10, 11,
                                 12, 13, 14, 15);
}

// CDNA5 async global->LDS copy (GVS mode: SGPR base + 32-bit VGPR offset).
// Tracked by ASYNCcnt; LDS dest address (wave-relative byte offset = low 32
// bits of the generic shared pointer) rides in a VGPR.
static __device__ __forceinline__ void async_b128(uint32_t lds, uint64_t base,
                                                  uint32_t off) {
  asm volatile("global_load_async_to_lds_b128 %0, %1, %2"
               :: "v"(lds), "v"(off), "s"(base)
               : "memory");
}

static __device__ __forceinline__ uint32_t lds_addr32(const void* p) {
  return (uint32_t)(uint64_t)p;
}

// ---------------------------------------------------------------------------
// Kernel 0: convert wq/wk/wv (f32, [C,W]) -> bf16 transposed [W,C] so that the
// WMMA B-matrix read is one contiguous 32B load per lane.
// ---------------------------------------------------------------------------
__global__ void k_wtrans(const float* __restrict__ wq, const float* __restrict__ wk,
                         const float* __restrict__ wv, __bf16* __restrict__ wT) {
  const int c = blockIdx.x * 16 + threadIdx.x;
  const int w = blockIdx.y * 16 + threadIdx.y;
  const int m = blockIdx.z;
  const float* src = (m == 0) ? wq : (m == 1) ? wk : wv;
  wT[(size_t)m * CC * WW + (size_t)w * CC + c] = (__bf16)src[(size_t)c * WW + w];
}

// ---------------------------------------------------------------------------
// Kernel 1: QKV projection.  Each wave computes one 16x16 output tile of
// X(BN x C) * W(C x W) + bias via 8 x v_wmma_f32_16x16x32_bf16.
// Q,K stored bf16 row-major [B,N,W]; V stored transposed bf16 [B,W,N].
// ---------------------------------------------------------------------------
__global__ void k_qkv(const float* __restrict__ x, const __bf16* __restrict__ wT,
                      const float* __restrict__ bq, const float* __restrict__ bk,
                      const float* __restrict__ bv, __bf16* __restrict__ Q,
                      __bf16* __restrict__ K, __bf16* __restrict__ Vt) {
  const int lane = threadIdx.x & 31;
  const int wave = threadIdx.x >> 5;
  const int half = lane >> 4;   // which 16-lane half
  const int l16  = lane & 15;
  const int mtile = blockIdx.x;               // 0 .. BN/16-1
  const int ntile = blockIdx.y * 4 + wave;    // 0 .. W/16-1
  const int mat   = blockIdx.z;               // 0=Q 1=K 2=V

  const __bf16* wTm  = wT + (size_t)mat * CC * WW;  // [W][C] bf16
  const float*  bias = (mat == 0) ? bq : (mat == 1) ? bk : bv;

  const int row0 = mtile * 16;
  const int ncol = ntile * 16 + l16;

  v8f acc = {};
  const float* xrow = x + (size_t)(row0 + l16) * CC;
#pragma unroll
  for (int c8 = 0; c8 < 8; ++c8) {
    // A operand (this code's contiguous k order, same as B): lane holds row M=l16, K spans
    // [half*16, half*16+16) within the 32-chunk.
    const int cb = c8 * 32 + half * 16;
    float4 f0 = *(const float4*)(xrow + cb);
    float4 f1 = *(const float4*)(xrow + cb + 4);
    float4 f2 = *(const float4*)(xrow + cb + 8);
    float4 f3 = *(const float4*)(xrow + cb + 12);
    v16bf a;
    a[0] = (__bf16)f0.x;  a[1] = (__bf16)f0.y;  a[2]  = (__bf16)f0.z;  a[3]  = (__bf16)f0.w;
    a[4] = (__bf16)f1.x;  a[5] = (__bf16)f1.y;  a[6]  = (__bf16)f1.z;  a[7]  = (__bf16)f1.w;
    a[8] = (__bf16)f2.x;  a[9] = (__bf16)f2.y;  a[10] = (__bf16)f2.z;  a[11] = (__bf16)f2.w;
    a[12] = (__bf16)f3.x; a[13] = (__bf16)f3.y; a[14] = (__bf16)f3.z;  a[15] = (__bf16)f3.w;
    // B-layout: lane holds column ncol, 16 consecutive K values starting at
    // chunk + half*16 -> one contiguous 32B load from transposed weights.
    v16bf bm = *(const v16bf*)(wTm + (size_t)ncol * CC + c8 * 32 + half * 16);
    acc = wmma_bf16(a, bm, acc);
  }
  const float bbv = bias[ncol];
  if (mat == 2) {
    // store V transposed: Vt[b][w][n]
    const int bidx = row0 / NN;
    const int nrow = (row0 % NN) + half * 8;
    __bf16* dst = Vt + ((size_t)bidx * WW + ncol) * NN + nrow;
#pragma unroll
    for (int r = 0; r < 8; ++r) dst[r] = (__bf16)(acc[r] + bbv);
  } else {
    __bf16* dst = (mat == 0) ? Q : K;
#pragma unroll
    for (int r = 0; r < 8; ++r)
      dst[(size_t)(row0 + half * 8 + r) * WW + ncol] = (__bf16)(acc[r] + bbv);
  }
}

// ---------------------------------------------------------------------------
// Kernel 2: flash attention.  Block = 4 waves, 64 query rows (16 per wave).
// Double-buffered key/value tiles streamed into LDS with
// global_load_async_to_lds_b128 (ASYNCcnt) so tile kt+1 loads overlap the
// 128 WMMAs of tile kt.  Per tile: S = Q*K^T (16 WMMA), masked online
// softmax, P through per-wave LDS (D-layout -> A-layout), O += P*V (16 WMMA).
// ---------------------------------------------------------------------------
__global__ void k_attn(const __bf16* __restrict__ Q, const __bf16* __restrict__ K,
                       const __bf16* __restrict__ Vt, const int* __restrict__ vlen,
                       float* __restrict__ out) {
  __shared__ __align__(16) __bf16 sK[2][32 * WW];    // 2 x 16KB
  __shared__ __align__(16) __bf16 sV[2][WW * 32];    // 2 x 16KB
  __shared__ __align__(16) __bf16 sP[4][16 * 32];    // per-wave P scratch (4KB)

  const int tid  = threadIdx.x;
  const int lane = tid & 31;
  const int wave = tid >> 5;
  const int half = lane >> 4;
  const int l16  = lane & 15;
  const int b    = blockIdx.y;
  const int qbase = blockIdx.x * 64 + wave * 16;
  const int valid = vlen[b];

  const __bf16* Kbase = K + (size_t)b * NN * WW;
  const __bf16* Vbase = Vt + (size_t)b * WW * NN;

  // Issue the 16 async b128 transfers that stage one 32-key tile (K: 16KB
  // contiguous; V: 256 rows x 64B) into LDS buffer `buf`.
  auto stage = [&](int kt, int buf) {
    const uint64_t kg = (uint64_t)(size_t)(Kbase + (size_t)kt * 32 * WW);
    const uint32_t klds = lds_addr32(&sK[buf][0]);
#pragma unroll
    for (int i = 0; i < 8; ++i) {
      const uint32_t off = (uint32_t)((tid + i * 128) * 16);
      async_b128(klds + off, kg, off);
    }
    const uint64_t vg = (uint64_t)(size_t)(Vbase + (size_t)kt * 32);
    const uint32_t vlds = lds_addr32(&sV[buf][0]);
#pragma unroll
    for (int r2 = 0; r2 < 2; ++r2) {
      const int wrow = tid * 2 + r2;
#pragma unroll
      for (int j = 0; j < 4; ++j)
        async_b128(vlds + (uint32_t)(wrow * 64 + j * 16), vg,
                   (uint32_t)(wrow * (NN * 2) + j * 16));
    }
  };

  // Q strip for this wave, A-layout, resident in registers (8 x v16bf).
  v16bf qa[8];
  {
    const __bf16* qrow = Q + ((size_t)b * NN + qbase + l16) * WW;
#pragma unroll
    for (int c = 0; c < 8; ++c) {
      v8bf lo = *(const v8bf*)(qrow + c * 32 + half * 16);
      v8bf hi = *(const v8bf*)(qrow + c * 32 + half * 16 + 8);
      qa[c] = cat8(lo, hi);
    }
  }

  v8f o[16];
#pragma unroll
  for (int t = 0; t < 16; ++t) o[t] = {};
  float mrow[8], lrow[8];
#pragma unroll
  for (int r = 0; r < 8; ++r) { mrow[r] = -3.0e38f; lrow[r] = 0.f; }

  stage(0, 0);  // prologue: tile 0 in flight

  for (int kt = 0; kt < NN / 32; ++kt) {
    const int cur = kt & 1;
    // Our 16 async ops for tile kt are the only outstanding ones.
    asm volatile("s_wait_asynccnt 0x0" ::: "memory");
    // All waves' loads landed AND all waves finished reading the other buffer.
    __syncthreads();
    if (kt + 1 < NN / 32) stage(kt + 1, cur ^ 1);  // overlap with compute

    const __bf16* sKc = sK[cur];
    const __bf16* sVc = sV[cur];

    // --- S = Q (16x256) * K_tile^T (256x32): two 16x16 D tiles ------------
    v8f s0 = {}, s1 = {};
#pragma unroll
    for (int c = 0; c < 8; ++c) {
      v16bf k0 = *(const v16bf*)(sKc + (size_t)l16 * WW + c * 32 + half * 16);
      v16bf k1 = *(const v16bf*)(sKc + (size_t)(16 + l16) * WW + c * 32 + half * 16);
      s0 = wmma_bf16(qa[c], k0, s0);
      s1 = wmma_bf16(qa[c], k1, s1);
    }

    // --- mask keys >= valid_len with -1e6 (reference semantics) -----------
    const int key0 = kt * 32 + l16;
    if (key0 >= valid) {
#pragma unroll
      for (int r = 0; r < 8; ++r) s0[r] = NEGV;
    }
    if (key0 + 16 >= valid) {
#pragma unroll
      for (int r = 0; r < 8; ++r) s1[r] = NEGV;
    }

    // --- online softmax (lane holds 8 rows of one column; reduce over the
    //     16 lanes of a half with xor-shuffles 1/2/4/8) ---------------------
    float scale[8];
#pragma unroll
    for (int r = 0; r < 8; ++r) {
      float t = fmaxf(s0[r], s1[r]);
      t = fmaxf(t, __shfl_xor(t, 1, 32));
      t = fmaxf(t, __shfl_xor(t, 2, 32));
      t = fmaxf(t, __shfl_xor(t, 4, 32));
      t = fmaxf(t, __shfl_xor(t, 8, 32));
      const float mn = fmaxf(mrow[r], t);
      scale[r] = __expf(mrow[r] - mn);
      s0[r] = __expf(s0[r] - mn);
      s1[r] = __expf(s1[r] - mn);
      float rs = s0[r] + s1[r];
      rs += __shfl_xor(rs, 1, 32);
      rs += __shfl_xor(rs, 2, 32);
      rs += __shfl_xor(rs, 4, 32);
      rs += __shfl_xor(rs, 8, 32);
      lrow[r] = lrow[r] * scale[r] + rs;
      mrow[r] = mn;
    }
#pragma unroll
    for (int t = 0; t < 16; ++t)
#pragma unroll
      for (int r = 0; r < 8; ++r) o[t][r] *= scale[r];

    // --- P: D-layout -> A-layout via per-wave LDS scratch ------------------
    __bf16* p = sP[wave];
#pragma unroll
    for (int r = 0; r < 8; ++r) {
      p[(half * 8 + r) * 32 + l16]      = (__bf16)s0[r];
      p[(half * 8 + r) * 32 + 16 + l16] = (__bf16)s1[r];
    }
    asm volatile("s_wait_dscnt 0" ::: "memory");  // intra-wave DS RAW fence
    v8bf plo = *(const v8bf*)(p + l16 * 32 + half * 16);
    v8bf phi = *(const v8bf*)(p + l16 * 32 + half * 16 + 8);
    v16bf pa = cat8(plo, phi);

    // --- O (16x256) += P (16x32) * V_tile (32x256) -------------------------
#pragma unroll
    for (int wt = 0; wt < 16; ++wt) {
      v16bf vb = *(const v16bf*)(sVc + (size_t)(wt * 16 + l16) * 32 + half * 16);
      o[wt] = wmma_bf16(pa, vb, o[wt]);
    }
  }

  // --- epilogue: divide by softmax denominator, store f32 ------------------
  float inv[8];
#pragma unroll
  for (int r = 0; r < 8; ++r) inv[r] = 1.f / lrow[r];
#pragma unroll
  for (int wt = 0; wt < 16; ++wt) {
#pragma unroll
    for (int r = 0; r < 8; ++r) {
      out[((size_t)b * NN + qbase + half * 8 + r) * WW + wt * 16 + l16] =
          o[wt][r] * inv[r];
    }
  }
}

// ---------------------------------------------------------------------------
extern "C" void kernel_launch(void* const* d_in, const int* in_sizes, int n_in,
                              void* d_out, int out_size, void* d_ws, size_t ws_size,
                              hipStream_t stream) {
  (void)in_sizes; (void)n_in; (void)out_size; (void)ws_size;
  const float* x    = (const float*)d_in[0];
  const int*   vlen = (const int*)d_in[1];
  const float* wq   = (const float*)d_in[2];
  const float* bq   = (const float*)d_in[3];
  const float* wk   = (const float*)d_in[4];
  const float* bk   = (const float*)d_in[5];
  const float* wv   = (const float*)d_in[6];
  const float* bv   = (const float*)d_in[7];
  float* out = (float*)d_out;

  // Workspace layout (bytes): wT (3*256*256*2 = 384KB), Q/K/Vt (16MB each).
  char* ws = (char*)d_ws;
  __bf16* wT = (__bf16*)ws;
  size_t off = (size_t)3 * CC * WW * 2;
  __bf16* Qb = (__bf16*)(ws + off);
  off += (size_t)BB * NN * WW * 2;
  __bf16* Kb = (__bf16*)(ws + off);
  off += (size_t)BB * NN * WW * 2;
  __bf16* Vtb = (__bf16*)(ws + off);

  k_wtrans<<<dim3(CC / 16, WW / 16, 3), dim3(16, 16), 0, stream>>>(wq, wk, wv, wT);
  k_qkv<<<dim3(BB * NN / 16, 4, 3), 128, 0, stream>>>(x, wT, bq, bk, bv, Qb, Kb, Vtb);
  k_attn<<<dim3(NN / 64, BB), 128, 0, stream>>>(Qb, Kb, Vtb, vlen, out);
}
